// BiRNN_CRF_26877905338536
// MI455X (gfx1250) — compile-verified, matrix operand order checked by reading, not run on hardware
//
#include <hip/hip_runtime.h>
#include <hip/hip_bf16.h>
#include <math.h>

// ---------- types ----------
typedef __bf16 v16bf __attribute__((ext_vector_type(16)));
typedef float  v8f   __attribute__((ext_vector_type(8)));
typedef unsigned short ushort_t;

union FragBF {
    uint4 q[2];   // two 16-byte chunks: K = kh..kh+7 and K = 16+kh..16+kh+7
    v16bf v;
};

// ---------- helpers ----------
__device__ __forceinline__ ushort_t f2bf(float f) {
    unsigned int u = __float_as_uint(f);
    u = (u + 0x7FFFu + ((u >> 16) & 1u)) >> 16;
    return (ushort_t)u;
}
__device__ __forceinline__ float bf2f(ushort_t b) {
    return __uint_as_float(((unsigned int)b) << 16);
}
__device__ __forceinline__ float sigmoidf(float x) {
    return 1.0f / (1.0f + __expf(-x));
}

// ---------- problem constants ----------
#define BB 32
#define LL 512
#define EE 1024
#define HH 512
#define G4 2048   // 4*H
#define H2 1024   // 2*H
#define TT 32

// ---------- workspace layout (bytes) ----------
#define OFF_XBF    ((size_t)0)
#define SZ_XBF     ((size_t)BB*LL*EE*2)            // 33,554,432
#define OFF_WIHF   (OFF_XBF + SZ_XBF)
#define SZ_WIH     ((size_t)G4*EE*2)               //  4,194,304
#define OFF_WIHB   (OFF_WIHF + SZ_WIH)
#define OFF_WHHF   (OFF_WIHB + SZ_WIH)
#define SZ_WHH     ((size_t)G4*HH*2)               //  2,097,152
#define OFF_WHHB   (OFF_WHHF + SZ_WHH)
#define OFF_WTAG   (OFF_WHHB + SZ_WHH)
#define SZ_WTAG    ((size_t)TT*H2*2)               //     65,536
#define OFF_XPF    (OFF_WTAG + SZ_WTAG)
#define SZ_XP      ((size_t)BB*LL*G4*4)            // 134,217,728
#define OFF_XPB    (OFF_XPF + SZ_XP)
#define OFF_HALL   (OFF_XPB + SZ_XP)
#define SZ_HALL    ((size_t)BB*LL*H2*2)            // 33,554,432
#define OFF_HNORM  (OFF_HALL + SZ_HALL)
#define OFF_EMIS   (OFF_HNORM + SZ_HALL)
#define SZ_EMIS    ((size_t)BB*LL*TT*4)            //  2,097,152

// ---------- fp32 -> bf16 conversion ----------
__global__ void cvt_bf16_kernel(const float* __restrict__ s, ushort_t* __restrict__ d, long n) {
    long i = (long)blockIdx.x * blockDim.x + threadIdx.x;
    long stride = (long)gridDim.x * blockDim.x;
    for (; i < n; i += stride) d[i] = f2bf(s[i]);
}

// ---------- generic WMMA GEMM:  out[M,N] = A[M,K] * B[N,K]^T + bias[N] ----------
// Each wave produces a 16 x (16*NTW) strip: the A fragment is loaded once per
// k-tile and reused across NTW B fragments (cuts b128 loads per WMMA from 4 to
// 2 + 2/NTW, raising arithmetic intensity toward the matrix-pipe roofline).
template<int NTW>
__global__ __launch_bounds__(256)
void wmma_gemm_nt_kernel(const ushort_t* __restrict__ A, const ushort_t* __restrict__ Bw,
                         const float* __restrict__ bias, float* __restrict__ out,
                         int M, int N, int K) {
    int wave = threadIdx.x >> 5;
    int lane = threadIdx.x & 31;
    int groups = (N >> 4) / NTW;                 // strip-groups per M-row
    long wt = (long)blockIdx.x * 8 + wave;
    int mt = (int)(wt / groups);
    int ng = (int)(wt % groups);
    if (mt >= (M >> 4)) return;
    int m0 = mt << 4;
    int n0 = ng * NTW * 16;
    int mlane = lane & 15;
    int kh = (lane >> 4) * 8;

    const ushort_t* arow = A + (size_t)(m0 + mlane) * K + kh;
    const ushort_t* brow[NTW];
    v8f acc[NTW];
    #pragma unroll
    for (int t = 0; t < NTW; ++t) {
        brow[t] = Bw + (size_t)(n0 + t * 16 + mlane) * K + kh;
        float bv = bias[n0 + t * 16 + mlane];
        #pragma unroll
        for (int i = 0; i < 8; ++i) acc[t][i] = bv;
    }

    for (int k0 = 0; k0 < K; k0 += 32) {
        FragBF a;
        a.q[0] = *(const uint4*)(arow + k0);
        a.q[1] = *(const uint4*)(arow + k0 + 16);
        __builtin_prefetch(arow + k0 + 128, 0, 1);   // global_prefetch_b8
        #pragma unroll
        for (int t = 0; t < NTW; ++t) {
            FragBF b;
            b.q[0] = *(const uint4*)(brow[t] + k0);
            b.q[1] = *(const uint4*)(brow[t] + k0 + 16);
            __builtin_prefetch(brow[t] + k0 + 128, 0, 1);
            acc[t] = __builtin_amdgcn_wmma_f32_16x16x32_bf16(false, a.v, false, b.v,
                                                             (short)0, acc[t], false, false);
        }
    }

    int mb = m0 + (lane >> 4) * 8;
    #pragma unroll
    for (int t = 0; t < NTW; ++t) {
        int nn = n0 + t * 16 + mlane;
        #pragma unroll
        for (int i = 0; i < 8; ++i) out[(size_t)(mb + i) * N + nn] = acc[t][i];
    }
}

// ---------- LSTM recurrence (both directions; batch split across blocks) ----------
// grid = 4 blocks of 512 threads (16 waves): blockIdx.x = dir*2 + mtile.
// Each block owns 16 batch rows of one direction. Each wave owns a 32-wide
// n-strip per gate quarter -> acc[4][2] = 64 VGPRs of accumulators.
// h state (16x512 bf16) lives in LDS; c state lives in registers; gates via WMMA.
// w_hh (2 MB/dir as bf16) is deliberately re-streamed from L2 every step: a
// laundered zero byte-offset makes the weight addresses formally loop-variant
// (so LICM cannot hoist the fragment loads out of the 512-step loop -> no
// scratch spills) while keeping pointer provenance on the kernel argument so
// the loads stay in the global address space (global_load_b128, not flat).
__global__ __launch_bounds__(512)
void lstm_kernel(const float* __restrict__ xpf, const float* __restrict__ xpb,
                 const ushort_t* __restrict__ whhf, const ushort_t* __restrict__ whhb,
                 ushort_t* __restrict__ hall) {
    __shared__ ushort_t hs[16 * HH];   // 16 KB

    int dir = blockIdx.x >> 1;
    int mt  = blockIdx.x & 1;
    const float*    xp  = dir ? xpb  : xpf;
    const ushort_t* whh = dir ? whhb : whhf;

    int wave  = threadIdx.x >> 5;   // 0..15
    int lane  = threadIdx.x & 31;
    int mlane = lane & 15;
    int half  = lane >> 4;
    int kh    = half * 8;
    int nlane = lane & 15;

    for (int i = threadIdx.x; i < 16 * HH; i += 512) hs[i] = 0;
    float creg[2][8];
    #pragma unroll
    for (int j = 0; j < 2; ++j)
        #pragma unroll
        for (int e = 0; e < 8; ++e) creg[j][e] = 0.0f;
    __syncthreads();

    size_t wzero = 0;   // stays 0, but the compiler must assume it changes per step

    for (int step = 0; step < LL; ++step) {
        int t = dir ? (LL - 1 - step) : step;

        // Launder a zero offset: defeats LICM of the weight loads without
        // destroying the global address-space provenance of `whh`.
        asm volatile("" : "+s"(wzero));
        const ushort_t* whw = whh + wzero;

        // init accumulators with x-projection (already contains bias)
        v8f acc[4][2];
        #pragma unroll
        for (int q = 0; q < 4; ++q)
            #pragma unroll
            for (int j = 0; j < 2; ++j) {
                int n = q * HH + wave * 32 + j * 16 + nlane;
                #pragma unroll
                for (int e = 0; e < 8; ++e) {
                    int m = e + half * 8;
                    int b = mt * 16 + m;
                    acc[q][j][e] = xp[((size_t)(b * LL + t)) * G4 + n];
                }
            }

        // gates += h @ w_hh^T   (K = 512)
        for (int kt = 0; kt < 16; ++kt) {
            int k0 = kt * 32;
            FragBF a;
            a.q[0] = *(const uint4*)&hs[mlane * HH + k0 + kh];
            a.q[1] = *(const uint4*)&hs[mlane * HH + k0 + kh + 16];
            #pragma unroll
            for (int q = 0; q < 4; ++q)
                #pragma unroll
                for (int j = 0; j < 2; ++j) {
                    int n = q * HH + wave * 32 + j * 16 + nlane;
                    const ushort_t* br = whw + (size_t)n * HH + k0 + kh;
                    FragBF bfr;
                    bfr.q[0] = *(const uint4*)br;
                    bfr.q[1] = *(const uint4*)(br + 16);
                    acc[q][j] = __builtin_amdgcn_wmma_f32_16x16x32_bf16(
                        false, a.v, false, bfr.v, (short)0, acc[q][j], false, false);
                }
        }

        // elementwise LSTM cell: i,f,g,o live in the four q-quarters
        ushort_t hnew[2][8];
        #pragma unroll
        for (int j = 0; j < 2; ++j) {
            #pragma unroll
            for (int e = 0; e < 8; ++e) {
                float ig = sigmoidf(acc[0][j][e]);
                float fg = sigmoidf(acc[1][j][e]);
                float gg = tanhf(acc[2][j][e]);
                float og = sigmoidf(acc[3][j][e]);
                float cc = fg * creg[j][e] + ig * gg;
                creg[j][e] = cc;
                float hh = og * tanhf(cc);
                hnew[j][e] = f2bf(hh);
                int m = e + half * 8;
                int b = mt * 16 + m;
                int n = wave * 32 + j * 16 + nlane;
                hall[((size_t)(b * LL + t)) * H2 + dir * HH + n] = hnew[j][e];
            }
        }
        __syncthreads();   // all GEMM reads of hs done
        #pragma unroll
        for (int j = 0; j < 2; ++j)
            #pragma unroll
            for (int e = 0; e < 8; ++e) {
                int m = e + half * 8;
                int n = wave * 32 + j * 16 + nlane;
                hs[m * HH + n] = hnew[j][e];
            }
        __syncthreads();   // hs updated for next step
    }
}

// ---------- LayerNorm over 2H (=1024), one block per (b,l) row ----------
__global__ __launch_bounds__(256)
void ln_kernel(const ushort_t* __restrict__ hall, const float* __restrict__ g,
               const float* __restrict__ bta, ushort_t* __restrict__ hnorm) {
    __shared__ float red[256];
    int r = blockIdx.x;
    const ushort_t* hrow = hall + (size_t)r * H2;
    float v[4];
    float s = 0.0f;
    #pragma unroll
    for (int i = 0; i < 4; ++i) { v[i] = bf2f(hrow[threadIdx.x * 4 + i]); s += v[i]; }
    red[threadIdx.x] = s; __syncthreads();
    for (int o = 128; o; o >>= 1) { if (threadIdx.x < o) red[threadIdx.x] += red[threadIdx.x + o]; __syncthreads(); }
    float mu = red[0] * (1.0f / H2);
    __syncthreads();
    float s2 = 0.0f;
    #pragma unroll
    for (int i = 0; i < 4; ++i) { float d = v[i] - mu; s2 += d * d; }
    red[threadIdx.x] = s2; __syncthreads();
    for (int o = 128; o; o >>= 1) { if (threadIdx.x < o) red[threadIdx.x] += red[threadIdx.x + o]; __syncthreads(); }
    float var = red[0] * (1.0f / H2);
    float rs = rsqrtf(var + 1e-5f);
    ushort_t* orow = hnorm + (size_t)r * H2;
    #pragma unroll
    for (int i = 0; i < 4; ++i) {
        int c = threadIdx.x * 4 + i;
        orow[c] = f2bf((v[i] - mu) * rs * g[c] + bta[c]);
    }
}

// ---------- CRF log-likelihood: 1 block, one wave32 per batch element ----------
__global__ __launch_bounds__(1024)
void crf_kernel(const float* __restrict__ emis, const int* __restrict__ tags,
                const int* __restrict__ mask, const float* __restrict__ trans,
                const float* __restrict__ startt, const float* __restrict__ endt,
                float* __restrict__ out) {
    __shared__ float alpha[BB][TT + 1];
    __shared__ float tr[TT][TT];
    __shared__ float llh[BB];

    int b = threadIdx.x >> 5;   // batch element (wave index)
    int j = threadIdx.x & 31;   // tag index (lane)

    tr[b][j] = trans[b * TT + j];
    __syncthreads();

    const float* eb = emis + (size_t)b * LL * TT;
    const int*   tg = tags + b * LL;
    const int*   mk = mask + b * LL;

    alpha[b][j] = startt[j] + eb[j];

    // numerator: lane-parallel over sequence positions
    float part = 0.0f;
    int mcount = 0;
    for (int l = j; l < LL; l += 32) mcount += (mk[l] != 0);
    for (int l = 1 + j; l < LL; l += 32) {
        float c = tr[tg[l - 1]][tg[l]] + eb[(size_t)l * TT + tg[l]];
        part += mk[l] ? c : 0.0f;
    }
    for (int o = 16; o; o >>= 1) {
        part   += __shfl_down(part, o, 32);
        mcount += __shfl_down(mcount, o, 32);
    }
    float num = 0.0f;
    if (j == 0) {
        num = part + startt[tg[0]] + eb[tg[0]];
        int last = mcount - 1;
        num += endt[tg[last]];
    }

    // forward algorithm (partition function)
    __syncthreads();
    for (int l = 1; l < LL; ++l) {
        float e = eb[(size_t)l * TT + j];
        float aold = alpha[b][j];
        float mx = -3.4e38f;
        #pragma unroll
        for (int i = 0; i < TT; ++i) mx = fmaxf(mx, alpha[b][i] + tr[i][j]);
        float s = 0.0f;
        #pragma unroll
        for (int i = 0; i < TT; ++i) s += __expf(alpha[b][i] + tr[i][j] - mx);
        float nxt = mx + __logf(s) + e;
        float anew = mk[l] ? nxt : aold;
        __syncthreads();
        alpha[b][j] = anew;
        __syncthreads();
    }

    // logZ = logsumexp(alpha + end_t)
    float v = alpha[b][j] + endt[j];
    float mx = v;
    for (int o = 16; o; o >>= 1) mx = fmaxf(mx, __shfl_xor(mx, o, 32));
    float se = __expf(v - mx);
    for (int o = 16; o; o >>= 1) se += __shfl_xor(se, o, 32);
    if (j == 0) llh[b] = num - (mx + __logf(se));
    __syncthreads();
    if (threadIdx.x == 0) {
        float s = 0.0f;
        for (int i = 0; i < BB; ++i) s += llh[i];
        out[0] = -s / (float)BB;
    }
}

// ---------- host launcher ----------
extern "C" void kernel_launch(void* const* d_in, const int* in_sizes, int n_in,
                              void* d_out, int out_size, void* d_ws, size_t ws_size,
                              hipStream_t stream) {
    const float* x      = (const float*)d_in[0];
    const int*   tags   = (const int*)d_in[1];
    const int*   amask  = (const int*)d_in[2];
    const float* w_ih_f = (const float*)d_in[3];
    const float* w_hh_f = (const float*)d_in[4];
    const float* b_f    = (const float*)d_in[5];
    const float* w_ih_b = (const float*)d_in[6];
    const float* w_hh_b = (const float*)d_in[7];
    const float* b_b    = (const float*)d_in[8];
    const float* ln_g   = (const float*)d_in[9];
    const float* ln_b   = (const float*)d_in[10];
    const float* W_tag  = (const float*)d_in[11];
    const float* b_tag  = (const float*)d_in[12];
    const float* trans  = (const float*)d_in[13];
    const float* startt = (const float*)d_in[14];
    const float* endt   = (const float*)d_in[15];

    char* ws = (char*)d_ws;
    ushort_t* xbf    = (ushort_t*)(ws + OFF_XBF);
    ushort_t* wihf   = (ushort_t*)(ws + OFF_WIHF);
    ushort_t* wihb   = (ushort_t*)(ws + OFF_WIHB);
    ushort_t* whhf   = (ushort_t*)(ws + OFF_WHHF);
    ushort_t* whhb   = (ushort_t*)(ws + OFF_WHHB);
    ushort_t* wtag   = (ushort_t*)(ws + OFF_WTAG);
    float*    xpf    = (float*)   (ws + OFF_XPF);
    float*    xpb    = (float*)   (ws + OFF_XPB);
    ushort_t* hall   = (ushort_t*)(ws + OFF_HALL);
    ushort_t* hnorm  = (ushort_t*)(ws + OFF_HNORM);
    float*    emis   = (float*)   (ws + OFF_EMIS);

    // 1) convert fp32 operands to bf16
    cvt_bf16_kernel<<<4096, 256, 0, stream>>>(x,      xbf,  (long)BB * LL * EE);
    cvt_bf16_kernel<<<2048, 256, 0, stream>>>(w_ih_f, wihf, (long)G4 * EE);
    cvt_bf16_kernel<<<2048, 256, 0, stream>>>(w_ih_b, wihb, (long)G4 * EE);
    cvt_bf16_kernel<<<1024, 256, 0, stream>>>(w_hh_f, whhf, (long)G4 * HH);
    cvt_bf16_kernel<<<1024, 256, 0, stream>>>(w_hh_b, whhb, (long)G4 * HH);
    cvt_bf16_kernel<<<64,   256, 0, stream>>>(W_tag,  wtag, (long)TT * H2);

    // 2) input projections: xproj = x * w_ih^T + b   (M=16384, N=2048, K=1024)
    //    16x64 strips per wave -> (1024 m-tiles) * (32 strip-groups) / 8 waves = 4096 blocks
    {
        int M = BB * LL, N = G4, K = EE;
        int blocks = (M >> 4) * ((N >> 4) / 4) / 8;   // 4096
        wmma_gemm_nt_kernel<4><<<blocks, 256, 0, stream>>>(xbf, wihf, b_f, xpf, M, N, K);
        wmma_gemm_nt_kernel<4><<<blocks, 256, 0, stream>>>(xbf, wihb, b_b, xpb, M, N, K);
    }

    // 3) bidirectional LSTM recurrence (4 blocks x 512 threads = 16 waves each)
    lstm_kernel<<<4, 512, 0, stream>>>(xpf, xpb, whhf, whhb, hall);

    // 4) LayerNorm
    ln_kernel<<<BB * LL, 256, 0, stream>>>(hall, ln_g, ln_b, hnorm);

    // 5) emissions = h_norm * W_tag^T + b_tag   (M=16384, N=32, K=1024)
    //    16x32 strips per wave -> 1024 * 1 / 8 = 128 blocks
    {
        int M = BB * LL, N = TT, K = H2;
        int blocks = (M >> 4) * ((N >> 4) / 2) / 8;   // 128
        wmma_gemm_nt_kernel<2><<<blocks, 256, 0, stream>>>(hnorm, wtag, b_tag, emis, M, N, K);
    }

    // 6) CRF log-likelihood -> scalar output
    crf_kernel<<<1, 1024, 0, stream>>>(emis, tags, amask, trans, startt, endt, (float*)d_out);
}
